// NeuronAstrocyteDQN_5583457485438
// MI455X (gfx1250) — compile-verified, matrix operand order checked by reading, not run on hardware
//
#include <hip/hip_runtime.h>

// ---------------------------------------------------------------------------
// CDNA5 WMMA types (wave32): v_wmma_f32_16x16x32_bf16
// ---------------------------------------------------------------------------
typedef __attribute__((ext_vector_type(16))) __bf16 v16bf;
typedef __attribute__((ext_vector_type(8)))  __bf16 v8bf;
typedef __attribute__((ext_vector_type(8)))  float  v8f;

__device__ __forceinline__ unsigned short f2bf_bits(float f) {
  // round-to-nearest-even fp32 -> bf16 (conversion / store paths only,
  // never in the WMMA inner loop)
  unsigned u = __builtin_bit_cast(unsigned, f);
  unsigned r = u + 0x7FFFu + ((u >> 16) & 1u);
  return (unsigned short)(r >> 16);
}

__device__ __forceinline__ float fast_tanh(float x) {
#if defined(__has_builtin)
#if __has_builtin(__builtin_amdgcn_tanhf)
  return __builtin_amdgcn_tanhf(x);   // v_tanh_f32 (CDNA5 trans op)
#else
  return tanhf(x);
#endif
#else
  return tanhf(x);
#endif
}

// ---------------------------------------------------------------------------
// One-time fp32 -> bf16 conversion (grid-stride, bandwidth bound)
// ---------------------------------------------------------------------------
__global__ void f32_to_bf16_kernel(const float* __restrict__ in,
                                   unsigned short* __restrict__ out, int n) {
  for (int i = blockIdx.x * blockDim.x + threadIdx.x; i < n;
       i += gridDim.x * blockDim.x)
    out[i] = f2bf_bits(in[i]);
}

// ---------------------------------------------------------------------------
// GEMM: C[M x N] = act(A[M x K] @ W[N x K]^T + bias[N]), A/W already bf16.
// One wave per 16x16 tile.  Per-lane fragment = 8 contiguous bf16 at two
// K-offsets (ISA 7.12.2 16-bit layout); every K is a multiple of 64 so all
// v8bf loads are 16B-aligned -> global_load_b128.
//
// N-edge handling (final layer N=18): out-of-range B lanes load a *clamped*
// valid W row instead of zero.  Column n of D depends only on column n of B,
// and out-of-range columns are never stored, so no zero-fill / exec-mask
// branching is needed in the inner loop.
//
// Two accumulators break the D->C dependency chain between v_wmma's.
// Cf (fp32 out) and/or Cb (bf16 out) may be null.
// ---------------------------------------------------------------------------
__global__ __launch_bounds__(32) void gemm_bf16_wmma(
    const __bf16* __restrict__ A, const __bf16* __restrict__ W,
    const float* __restrict__ bias, float* __restrict__ Cf,
    unsigned short* __restrict__ Cb, int M, int N, int K, int relu)
{
  const int tilesN = (N + 15) >> 4;
  const int tile = blockIdx.x;
  const int m0 = (tile / tilesN) << 4;
  const int n0 = (tile % tilesN) << 4;
  const int lane = threadIdx.x & 31;
  const int l15  = lane & 15;
  const int hi8  = (lane >> 4) << 3;       // 0 or 8

  const int rA  = m0 + l15;                // M is always a multiple of 16
  const int rB  = n0 + l15;                // may exceed N (final layer N=18)
  const bool bok = (rB < N);
  const int rBc = bok ? rB : (N - 1);      // clamped: loads stay in-bounds

  v8f acc0, acc1;
  {
    const float bv = bok ? bias[rB] : 0.0f;
    #pragma unroll
    for (int t = 0; t < 8; ++t) { acc0[t] = bv; acc1[t] = 0.0f; }
  }

  const __bf16* arow = A + (long)rA  * K;
  const __bf16* brow = W + (long)rBc * K;

  union Frag { v16bf v; v8bf h[2]; };

  for (int k0 = 0; k0 < K; k0 += 64) {
    Frag a0, a1, b0, b1;
    a0.h[0] = *(const v8bf*)(arow + k0 + hi8);
    a0.h[1] = *(const v8bf*)(arow + k0 + 16 + hi8);
    a1.h[0] = *(const v8bf*)(arow + k0 + 32 + hi8);
    a1.h[1] = *(const v8bf*)(arow + k0 + 48 + hi8);
    b0.h[0] = *(const v8bf*)(brow + k0 + hi8);
    b0.h[1] = *(const v8bf*)(brow + k0 + 16 + hi8);
    b1.h[0] = *(const v8bf*)(brow + k0 + 32 + hi8);
    b1.h[1] = *(const v8bf*)(brow + k0 + 48 + hi8);
    acc0 = __builtin_amdgcn_wmma_f32_16x16x32_bf16(
        false, a0.v, false, b0.v, (short)0, acc0, false, false);
    acc1 = __builtin_amdgcn_wmma_f32_16x16x32_bf16(
        false, a1.v, false, b1.v, (short)0, acc1, false, false);
  }

  const int col = n0 + l15;
  const int rowbase = m0 + ((lane >> 4) ? 8 : 0);
  if (col < N) {
    #pragma unroll
    for (int r = 0; r < 8; ++r) {
      float v = acc0[r] + acc1[r];
      if (relu) v = fmaxf(v, 0.0f);
      if (Cf) Cf[(long)(rowbase + r) * N + col] = v;
      if (Cb) Cb[(long)(rowbase + r) * N + col] = f2bf_bits(v);
    }
  }
}

// ---------------------------------------------------------------------------
// Hopfield-like retrieval: one workgroup per batch sample, S/P resident in
// LDS for all iterations (~140KB of the 320KB WGP LDS).
// ---------------------------------------------------------------------------
#define NE 128
#define KP 10
#define BETA_ 5.0f
#define DT_   0.05f

__global__ __launch_bounds__(256) void retrieve_kernel(
    const float* __restrict__ enc, const float* __restrict__ etas,
    float* __restrict__ out)
{
  __shared__ float S[NE * NE];
  __shared__ float P[NE * NE];
  __shared__ float h[NE];
  __shared__ float xx[NE];
  __shared__ float gh[NE];
  __shared__ float tt[NE * KP];
  __shared__ float cc[KP];
  __shared__ float et[NE * KP];

  const int tid = threadIdx.x;
  const int b   = blockIdx.x;
  const float invN3 = 1.0f / (128.0f * 128.0f * 128.0f);

  for (int i = tid; i < NE * KP; i += 256) et[i] = etas[i];

  if (tid < NE) {
    const float e  = enc[b * NE + tid];
    const float x0 = (e > 0.0f) ? 1.0f : -1.0f;   // sign, zeros -> -1
    xx[tid] = x0;
    h[tid]  = fast_tanh(BETA_ * x0);
  }
  __syncthreads();

  // c0[m] = -(h0 . eta_m)^2   (closed form of the quartic on psi0 = -h0 h0^T)
  if (tid < KP) {
    float d = 0.0f;
    for (int l = 0; l < NE; ++l) d += h[l] * et[l * KP + tid];
    cc[tid] = -d * d;
  }
  __syncthreads();

  // P0 = atanh(-h0 h0^T)/B ;  S0 = atanh(-quart0/N^3)/B
  for (int idx = tid; idx < NE * NE; idx += 256) {
    const int i = idx >> 7, j = idx & 127;
    P[idx] = atanhf(-h[i] * h[j]) * (1.0f / BETA_);
    float q = 0.0f;
    #pragma unroll
    for (int m = 0; m < KP; ++m) q += et[i * KP + m] * cc[m] * et[j * KP + m];
    S[idx] = atanhf(-q * invN3) * (1.0f / BETA_);
  }
  __syncthreads();

  // 99 full update steps; output is tanh(B*xx) at the top of step 100.
  for (int it = 0; it < 99; ++it) {
    if (tid < NE) h[tid] = fast_tanh(BETA_ * xx[tid]);
    __syncthreads();

    if (tid < NE) {
      // Row tid is private to this thread: fuse g/psi, gh, t = psi@eta, and
      // the in-place S update + non-quartic part of the P update.
      const float hi_ = h[tid];
      float acc_gh = 0.0f;
      float tm[KP];
      #pragma unroll
      for (int m = 0; m < KP; ++m) tm[m] = 0.0f;
      for (int j = 0; j < NE; ++j) {
        const int idx = tid * NE + j;
        const float s_old = S[idx];
        const float p_old = P[idx];
        const float g   = fast_tanh(BETA_ * s_old);
        const float psi = fast_tanh(BETA_ * p_old);
        acc_gh += g * h[j];
        #pragma unroll
        for (int m = 0; m < KP; ++m) tm[m] += psi * et[j * KP + m];
        S[idx] = s_old + DT_ * (-s_old + hi_ * h[j] + psi);
        P[idx] = p_old + DT_ * (-p_old + g);          // + DT*quart/N^3 below
      }
      gh[tid] = acc_gh;
      #pragma unroll
      for (int m = 0; m < KP; ++m) tt[tid * KP + m] = tm[m];
    }
    __syncthreads();

    if (tid < KP) {
      float c = 0.0f;
      for (int k = 0; k < NE; ++k) c += et[k * KP + tid] * tt[k * KP + tid];
      cc[tid] = c;
    }
    if (tid < NE) xx[tid] += DT_ * (-xx[tid] + gh[tid]);
    __syncthreads();

    for (int idx = tid; idx < NE * NE; idx += 256) {
      const int i = idx >> 7, j = idx & 127;
      float q = 0.0f;
      #pragma unroll
      for (int m = 0; m < KP; ++m) q += et[i * KP + m] * cc[m] * et[j * KP + m];
      P[idx] += (DT_ * invN3) * q;
    }
    __syncthreads();
  }

  if (tid < NE) out[b * NE + tid] = fast_tanh(BETA_ * xx[tid]);
}

// ---------------------------------------------------------------------------
// combined = concat(x, retrieved) written directly as bf16 [128 x 1152]
// ---------------------------------------------------------------------------
__global__ void concat_bf16_kernel(const float* __restrict__ x,
                                   const float* __restrict__ ret,
                                   unsigned short* __restrict__ comb)
{
  const int idx = blockIdx.x * blockDim.x + threadIdx.x;
  if (idx >= 128 * 1152) return;
  const int b = idx / 1152, k = idx - b * 1152;
  const float v = (k < 1024) ? x[b * 1024 + k] : ret[b * 128 + (k - 1024)];
  comb[idx] = f2bf_bits(v);
}

// ---------------------------------------------------------------------------
// Launch
// in:  0 x(128x1024) 1 etas(128x10) 2 We1(2048x1024) 3 be1 4 We2(128x2048)
//      5 be2 6 Wn1(2048x1152) 7 bn1 8 Wn2(2048x2048) 9 bn2 10 Wn3(18x2048) 11 bn3
// out: (128x18) fp32
// ---------------------------------------------------------------------------
extern "C" void kernel_launch(void* const* d_in, const int* in_sizes, int n_in,
                              void* d_out, int out_size, void* d_ws, size_t ws_size,
                              hipStream_t stream) {
  (void)in_sizes; (void)n_in; (void)out_size; (void)ws_size;
  const float* x    = (const float*)d_in[0];
  const float* etas = (const float*)d_in[1];
  const float* We1  = (const float*)d_in[2];
  const float* be1  = (const float*)d_in[3];
  const float* We2  = (const float*)d_in[4];
  const float* be2  = (const float*)d_in[5];
  const float* Wn1  = (const float*)d_in[6];
  const float* bn1  = (const float*)d_in[7];
  const float* Wn2  = (const float*)d_in[8];
  const float* bn2  = (const float*)d_in[9];
  const float* Wn3  = (const float*)d_in[10];
  const float* bn3  = (const float*)d_in[11];
  float* out = (float*)d_out;

  // fp32 scratch
  float* wsf  = (float*)d_ws;
  float* enc  = wsf;  wsf += 128 * 128;
  float* retr = wsf;  wsf += 128 * 128;
  // bf16 scratch (after fp32 region)
  unsigned short* wsb = (unsigned short*)wsf;
  unsigned short* xb   = wsb;  wsb += 128 * 1024;
  unsigned short* We1b = wsb;  wsb += 2048 * 1024;
  unsigned short* We2b = wsb;  wsb += 128 * 2048;
  unsigned short* Wn1b = wsb;  wsb += 2048 * 1152;
  unsigned short* Wn2b = wsb;  wsb += 2048 * 2048;
  unsigned short* Wn3b = wsb;  wsb += 18 * 2048;
  unsigned short* h1eb = wsb;  wsb += 128 * 2048;
  unsigned short* combb= wsb;  wsb += 128 * 1152;
  unsigned short* h1b  = wsb;  wsb += 128 * 2048;
  unsigned short* h2b  = wsb;  /* += 128 * 2048 */

  const int M = 128;
  const dim3 cb(256);
  auto cvt = [&](const float* src, unsigned short* dst, int n) {
    f32_to_bf16_kernel<<<dim3((n + 255) / 256), cb, 0, stream>>>(src, dst, n);
  };
  // one-time fp32 -> bf16 for all GEMM operands that arrive in fp32
  cvt(x,   xb,   128 * 1024);
  cvt(We1, We1b, 2048 * 1024);
  cvt(We2, We2b, 128 * 2048);
  cvt(Wn1, Wn1b, 2048 * 1152);
  cvt(Wn2, Wn2b, 2048 * 2048);
  cvt(Wn3, Wn3b, 18 * 2048);

  // encoder layer 1: h1e(bf16) = relu(x @ We1^T + be1)   [128 x 2048], K=1024
  gemm_bf16_wmma<<<dim3((M/16) * (2048/16)), dim3(32), 0, stream>>>(
      (const __bf16*)xb, (const __bf16*)We1b, be1, nullptr, h1eb,
      M, 2048, 1024, 1);
  // encoder layer 2: enc(fp32) = h1e @ We2^T + be2       [128 x 128], K=2048
  gemm_bf16_wmma<<<dim3((M/16) * (128/16)), dim3(32), 0, stream>>>(
      (const __bf16*)h1eb, (const __bf16*)We2b, be2, enc, nullptr,
      M, 128, 2048, 0);

  // per-sample Hopfield retrieval (LDS-resident state)
  retrieve_kernel<<<dim3(128), dim3(256), 0, stream>>>(enc, etas, retr);

  // concat -> bf16 combined [128 x 1152]
  concat_bf16_kernel<<<dim3((128 * 1152 + 255) / 256), cb, 0, stream>>>(
      x, retr, combb);

  // decoder NN (bf16 activations throughout)
  gemm_bf16_wmma<<<dim3((M/16) * (2048/16)), dim3(32), 0, stream>>>(
      (const __bf16*)combb, (const __bf16*)Wn1b, bn1, nullptr, h1b,
      M, 2048, 1152, 1);
  gemm_bf16_wmma<<<dim3((M/16) * (2048/16)), dim3(32), 0, stream>>>(
      (const __bf16*)h1b, (const __bf16*)Wn2b, bn2, nullptr, h2b,
      M, 2048, 2048, 1);
  // final: out(fp32) = h2 @ Wn3^T + bn3   [128 x 18], 2 column tiles, guarded
  gemm_bf16_wmma<<<dim3((M/16) * 2), dim3(32), 0, stream>>>(
      (const __bf16*)h2b, (const __bf16*)Wn3b, bn3, out, nullptr,
      M, 18, 2048, 0);
}
